// DiscreteComm_29214367547981
// MI455X (gfx1250) — compile-verified, matrix operand order checked by reading, not run on hardware
//
#include <hip/hip_runtime.h>
#include <hip/hip_bf16.h>

// ---------------------------------------------------------------------------
// DiscreteComm GNN step for MI455X (gfx1250), wave32 + WMMA f16->f32.
//
// Prep   : one-time f32->f16 conversion of x, h, W_enc, W_dec, W_ih, W_hh.
// Stage 1: edge logits via v_wmma_f32_16x16x32_f16; W_enc16 staged into 128KB
//          LDS with GLOBAL_LOAD_ASYNC_TO_LDS_B128 (ASYNCcnt path, no VGPR
//          round-trip); straight-through hard gumbel == pairwise compare;
//          pack 128-bit mask; atomicOr per dst node (segment-max == OR).
// Stage 2: dec = c @ W_dec^T, A fragments built from mask bits (0/1 exact).
// Stage 3: fused GRU: r/z/n GEMMs per 16-column slice + sigmoid/tanh epilogue.
// ---------------------------------------------------------------------------

#define NN 20000
#define NE 640000
#define HID 256
#define MSGB 128          // 2*MSG one-hot bits

typedef __attribute__((ext_vector_type(16))) _Float16 v16h;
typedef __attribute__((ext_vector_type(8)))  _Float16 v8h;
typedef __attribute__((ext_vector_type(4)))  _Float16 v4h;
typedef __attribute__((ext_vector_type(8)))  float    v8f;

// A/B fragment layout (16x16x32 f16): lane L holds row/col (L&15); the 16
// halves are K = K0..K0+7 and K0+16..K0+23 with K0 = (L<16 ? 0 : 8).
__device__ __forceinline__ v16h frag_f16(const _Float16* __restrict__ p0,
                                         const _Float16* __restrict__ p1) {
  v8h a = *(const v8h*)p0;
  v8h b = *(const v8h*)p1;
  v16h f;
#pragma unroll
  for (int i = 0; i < 8; ++i) { f[i] = a[i]; f[i + 8] = b[i]; }
  return f;
}

__device__ __forceinline__ float sigm(float v) { return 1.0f / (1.0f + __expf(-v)); }

// Async 16B global->LDS copy (VGLOBAL GV mode, tracked by ASYNCcnt).
__device__ __forceinline__ void async_copy_b128(const void* gaddr, void* lds) {
  unsigned ldsOff = (unsigned)(size_t)lds;            // LDS byte address
  unsigned long long ga = (unsigned long long)(size_t)gaddr;
  asm volatile("global_load_async_to_lds_b128 %0, %1, off"
               :: "v"(ldsOff), "v"(ga) : "memory");
}
__device__ __forceinline__ void async_drain() {
  asm volatile("s_wait_asynccnt 0x0" ::: "memory");
}

// ---------------------------------------------------------------------------
__global__ void zero_mask_kernel(unsigned* __restrict__ p, int n) {
  int i = blockIdx.x * blockDim.x + threadIdx.x;
  if (i < n) p[i] = 0u;
}

// Vectorized one-time f32 -> f16 conversion (n % 4 == 0 for all operands).
__global__ void cvt_f16_kernel(const float* __restrict__ src,
                               _Float16* __restrict__ dst, int n4) {
  int i = blockIdx.x * blockDim.x + threadIdx.x;
  if (i < n4) {
    float4 w = *(const float4*)(src + (size_t)i * 4);
    v4h hv; hv[0]=(_Float16)w.x; hv[1]=(_Float16)w.y; hv[2]=(_Float16)w.z; hv[3]=(_Float16)w.w;
    *(v4h*)(dst + (size_t)i * 4) = hv;
  }
}

// ---------------------------------------------------------------------------
// Stage 1: per-wave 16 edges x 128 logits, K = 512 (x16[src] || h16[src]).
__global__ __launch_bounds__(256) void edge_msg_kernel(
    const _Float16* __restrict__ x16, const _Float16* __restrict__ h16,
    const _Float16* __restrict__ Wenc16, const float* __restrict__ b_enc,
    const float* __restrict__ gumbel,
    const int* __restrict__ src, const int* __restrict__ dst,
    unsigned* __restrict__ cmask) {
  __shared__ _Float16 sW[MSGB * 2 * HID];   // 128 x 512 f16 = 128 KB LDS

  // Async global->LDS stage of W_enc16 (no VGPR round-trip, ASYNCcnt path).
#pragma unroll 1
  for (int i = threadIdx.x * 8; i < MSGB * 2 * HID; i += 256 * 8) {
    async_copy_b128(Wenc16 + i, &sW[i]);
  }
  async_drain();          // each wave drains its own ASYNCcnt ...
  __syncthreads();        // ... barrier publishes all waves' LDS writes

  const int lane  = threadIdx.x & 31;
  const int wave  = threadIdx.x >> 5;
  const int ebase = (blockIdx.x * 8 + wave) * 16;     // NE % 128 == 0
  const int nl    = lane & 15;
  const int K0    = (lane & 16) ? 8 : 0;

  const int e = ebase + nl;                 // A row for this lane
  const int s = src[e];
  const _Float16* xrow = x16 + (size_t)s * HID;
  const _Float16* hrow = h16 + (size_t)s * HID;

  v8f acc[8] = {};
#pragma unroll 4
  for (int kk = 0; kk < 16; ++kk) {
    const int kb = kk * 32;
    const _Float16* rowp = (kb < HID) ? xrow : hrow;
    const int ko = (kb < HID) ? kb : (kb - HID);
    v16h A = frag_f16(rowp + ko + K0, rowp + ko + K0 + 16);
#pragma unroll
    for (int t = 0; t < 8; ++t) {
      const _Float16* wr = &sW[(size_t)(t * 16 + nl) * (2 * HID) + kb + K0];
      v16h B = frag_f16(wr, wr + 16);
      acc[t] = __builtin_amdgcn_wmma_f32_16x16x32_f16(
          false, A, false, B, (short)0, acc[t], false, false);
    }
  }

  // Epilogue: logits+gumbel pairwise argmax -> ballot -> 128-bit mask -> OR.
  const int mg = (lane & 16) ? 8 : 0;       // C row group for this lane
  const int d  = dst[e];                    // used by lanes 0..15 (row = nl)
#pragma unroll
  for (int w = 0; w < 4; ++w) {
    unsigned selw[2];
#pragma unroll
    for (int half = 0; half < 2; ++half) {
      const int t = 2 * w + half;
      const int n = t * 16 + nl;
      const float bia = b_enc[n];
      unsigned sel = 0;
#pragma unroll
      for (int v = 0; v < 8; ++v) {
        const int M = v + mg;
        const float g = gumbel[(size_t)(ebase + M) * MSGB + n];
        const float val = acc[t][v] + bia + g;
        const float pv  = __shfl_xor(val, 1, 32);
        const bool bit  = (n & 1) ? (val > pv) : (val >= pv);
        const unsigned bal = (unsigned)__ballot(bit);
        sel |= (nl == v)     ? (bal & 0xffffu) : 0u;
        sel |= (nl == v + 8) ? (bal >> 16)     : 0u;
      }
      selw[half] = sel;
    }
    if (lane < 16)
      atomicOr(&cmask[(size_t)d * 4 + w], selw[0] | (selw[1] << 16));
  }
}

// ---------------------------------------------------------------------------
// Stage 2: dec[n,128] = c @ W_dec^T + b_dec, c bits taken straight from mask.
__global__ __launch_bounds__(256) void dec_kernel(
    const unsigned* __restrict__ cmask,
    const _Float16* __restrict__ Wdec16, const float* __restrict__ b_dec,
    _Float16* __restrict__ decws) {
  __shared__ _Float16 sW[MSGB * MSGB];      // 128 x 128 f16 = 32 KB
#pragma unroll 1
  for (int i = threadIdx.x * 8; i < MSGB * MSGB; i += 256 * 8) {
    async_copy_b128(Wdec16 + i, &sW[i]);
  }
  async_drain();
  __syncthreads();

  const int lane  = threadIdx.x & 31;
  const int wave  = threadIdx.x >> 5;
  const int nbase = (blockIdx.x * 8 + wave) * 16;
  if (nbase >= NN) return;
  const int nl = lane & 15;
  const int K0 = (lane & 16) ? 8 : 0;

  const unsigned* mp = cmask + (size_t)(nbase + nl) * 4;
  unsigned mw[4] = {mp[0], mp[1], mp[2], mp[3]};

  v8f acc[8] = {};
#pragma unroll
  for (int kk = 0; kk < 4; ++kk) {
    const unsigned bits0 = (mw[kk] >> K0) & 0xffu;
    const unsigned bits1 = (mw[kk] >> (K0 + 16)) & 0xffu;
    v16h A;
#pragma unroll
    for (int i = 0; i < 8; ++i) {
      A[i]     = (_Float16)(float)((bits0 >> i) & 1u);
      A[i + 8] = (_Float16)(float)((bits1 >> i) & 1u);
    }
#pragma unroll
    for (int t = 0; t < 8; ++t) {
      const _Float16* wr = &sW[(size_t)(t * 16 + nl) * MSGB + kk * 32 + K0];
      v16h B = frag_f16(wr, wr + 16);
      acc[t] = __builtin_amdgcn_wmma_f32_16x16x32_f16(
          false, A, false, B, (short)0, acc[t], false, false);
    }
  }

  const int mg = (lane & 16) ? 8 : 0;
#pragma unroll
  for (int t = 0; t < 8; ++t) {
    const int o = t * 16 + nl;
    const float bd = b_dec[o];
#pragma unroll
    for (int v = 0; v < 8; ++v) {
      const int nd = nbase + v + mg;
      decws[(size_t)nd * MSGB + o] = (_Float16)(acc[t][v] + bd);
    }
  }
}

// ---------------------------------------------------------------------------
// Stage 3: fused GRU. Per wave: 16 nodes; loop over 16 column slices of 16.
// inp = [x (256) || dec (128)] -> W_ih rows; h -> W_hh rows. All f16 operands.
__global__ __launch_bounds__(256) void gru_kernel(
    const _Float16* __restrict__ x16, const _Float16* __restrict__ h16,
    const float* __restrict__ h, const _Float16* __restrict__ decws,
    const _Float16* __restrict__ Wih16, const float* __restrict__ b_ih,
    const _Float16* __restrict__ Whh16, const float* __restrict__ b_hh,
    float* __restrict__ out) {
  const int lane  = threadIdx.x & 31;
  const int wave  = threadIdx.x >> 5;
  const int nbase = (blockIdx.x * 8 + wave) * 16;
  if (nbase >= NN) return;
  const int nl = lane & 15;
  const int K0 = (lane & 16) ? 8 : 0;
  const int mg = (lane & 16) ? 8 : 0;

  const int ndA = nbase + nl;               // A row for this lane
  const _Float16* xrow = x16 + (size_t)ndA * HID;
  const _Float16* hrow = h16 + (size_t)ndA * HID;
  const _Float16* drow = decws + (size_t)ndA * MSGB;

#pragma unroll 1
  for (int ot = 0; ot < 16; ++ot) {
    const int o = ot * 16 + nl;             // output column within a gate
    v8f acc_r = {}, acc_z = {}, acc_ni = {}, acc_nh = {};

    // K-loop over inp = [x || dec], K = 384.
#pragma unroll 2
    for (int kk = 0; kk < 12; ++kk) {
      const int kb = kk * 32;
      v16h A;
      if (kb < HID) A = frag_f16(xrow + kb + K0, xrow + kb + K0 + 16);
      else          A = frag_f16(drow + (kb - HID) + K0, drow + (kb - HID) + K0 + 16);
      const _Float16* wr0 = Wih16 + (size_t)(o)         * 384 + kb + K0;
      const _Float16* wr1 = Wih16 + (size_t)(HID + o)   * 384 + kb + K0;
      const _Float16* wr2 = Wih16 + (size_t)(2*HID + o) * 384 + kb + K0;
      v16h B0 = frag_f16(wr0, wr0 + 16);
      v16h B1 = frag_f16(wr1, wr1 + 16);
      v16h B2 = frag_f16(wr2, wr2 + 16);
      acc_r  = __builtin_amdgcn_wmma_f32_16x16x32_f16(false, A, false, B0, (short)0, acc_r,  false, false);
      acc_z  = __builtin_amdgcn_wmma_f32_16x16x32_f16(false, A, false, B1, (short)0, acc_z,  false, false);
      acc_ni = __builtin_amdgcn_wmma_f32_16x16x32_f16(false, A, false, B2, (short)0, acc_ni, false, false);
    }
    // K-loop over h, K = 256.
#pragma unroll 2
    for (int kk = 0; kk < 8; ++kk) {
      const int kb = kk * 32;
      v16h A = frag_f16(hrow + kb + K0, hrow + kb + K0 + 16);
      const _Float16* wr0 = Whh16 + (size_t)(o)         * HID + kb + K0;
      const _Float16* wr1 = Whh16 + (size_t)(HID + o)   * HID + kb + K0;
      const _Float16* wr2 = Whh16 + (size_t)(2*HID + o) * HID + kb + K0;
      v16h B0 = frag_f16(wr0, wr0 + 16);
      v16h B1 = frag_f16(wr1, wr1 + 16);
      v16h B2 = frag_f16(wr2, wr2 + 16);
      acc_r  = __builtin_amdgcn_wmma_f32_16x16x32_f16(false, A, false, B0, (short)0, acc_r,  false, false);
      acc_z  = __builtin_amdgcn_wmma_f32_16x16x32_f16(false, A, false, B1, (short)0, acc_z,  false, false);
      acc_nh = __builtin_amdgcn_wmma_f32_16x16x32_f16(false, A, false, B2, (short)0, acc_nh, false, false);
    }

    // Biases (column-only dependence; hoisted over the 8 C rows).
    const float br = b_ih[o]         + b_hh[o];
    const float bz = b_ih[HID + o]   + b_hh[HID + o];
    const float bi = b_ih[2*HID + o];
    const float bh = b_hh[2*HID + o];

#pragma unroll
    for (int v = 0; v < 8; ++v) {
      const int nd = nbase + v + mg;        // C row -> node
      const float r  = sigm(acc_r[v] + br);
      const float z  = sigm(acc_z[v] + bz);
      const float hn = acc_nh[v] + bh;
      const float n  = tanhf(acc_ni[v] + bi + r * hn);
      const float hv = h[(size_t)nd * HID + o];   // exact f32 carry term
      out[(size_t)nd * HID + o] = (1.0f - z) * n + z * hv;
    }
  }
}

// ---------------------------------------------------------------------------
extern "C" void kernel_launch(void* const* d_in, const int* in_sizes, int n_in,
                              void* d_out, int out_size, void* d_ws, size_t ws_size,
                              hipStream_t stream) {
  const float* x      = (const float*)d_in[0];
  const float* h      = (const float*)d_in[1];
  const float* W_enc  = (const float*)d_in[2];
  const float* b_enc  = (const float*)d_in[3];
  const float* W_dec  = (const float*)d_in[4];
  const float* b_dec  = (const float*)d_in[5];
  const float* W_ih   = (const float*)d_in[6];
  const float* b_ih   = (const float*)d_in[7];
  const float* W_hh   = (const float*)d_in[8];
  const float* b_hh   = (const float*)d_in[9];
  const float* gumbel = (const float*)d_in[10];
  const int*   src    = (const int*)d_in[11];
  const int*   dst    = (const int*)d_in[12];
  float* out = (float*)d_out;

  // Workspace layout (bytes):
  char* ws = (char*)d_ws;
  unsigned*  cmask  = (unsigned*)ws;              ws += (size_t)NN * 4 * 4;      // 320 KB
  _Float16*  decws  = (_Float16*)ws;              ws += (size_t)NN * MSGB * 2;   // 5 MB
  _Float16*  x16    = (_Float16*)ws;              ws += (size_t)NN * HID * 2;    // 10 MB
  _Float16*  h16    = (_Float16*)ws;              ws += (size_t)NN * HID * 2;    // 10 MB
  _Float16*  Wenc16 = (_Float16*)ws;              ws += (size_t)MSGB * 2*HID*2;  // 128 KB
  _Float16*  Wdec16 = (_Float16*)ws;              ws += (size_t)MSGB * MSGB * 2; // 32 KB
  _Float16*  Wih16  = (_Float16*)ws;              ws += (size_t)3*HID * 384 * 2; // 576 KB
  _Float16*  Whh16  = (_Float16*)ws;              ws += (size_t)3*HID * HID * 2; // 384 KB

  const int mask_words = NN * 4;
  zero_mask_kernel<<<(mask_words + 255) / 256, 256, 0, stream>>>(cmask, mask_words);

  // One-time f32 -> f16 conversions (bandwidth-trivial vs the 84 GFLOP GEMM).
  auto cvt = [&](const float* s, _Float16* d, size_t n) {
    int n4 = (int)(n / 4);
    cvt_f16_kernel<<<(n4 + 255) / 256, 256, 0, stream>>>(s, d, n4);
  };
  cvt(x,     x16,    (size_t)NN * HID);
  cvt(h,     h16,    (size_t)NN * HID);
  cvt(W_enc, Wenc16, (size_t)MSGB * 2 * HID);
  cvt(W_dec, Wdec16, (size_t)MSGB * MSGB);
  cvt(W_ih,  Wih16,  (size_t)3 * HID * 384);
  cvt(W_hh,  Whh16,  (size_t)3 * HID * HID);

  edge_msg_kernel<<<NE / 128, 256, 0, stream>>>(x16, h16, Wenc16, b_enc, gumbel,
                                                src, dst, cmask);

  const int nblocks = (NN + 127) / 128;
  dec_kernel<<<nblocks, 256, 0, stream>>>(cmask, Wdec16, b_dec, decws);
  gru_kernel<<<nblocks, 256, 0, stream>>>(x16, h16, h, decws, Wih16, b_ih,
                                          Whh16, b_hh, out);
}
